// TimeConvX_17291538333994
// MI455X (gfx1250) — matrix operand order repaired and checked by reading, not compiled, not run on hardware
//
#include <hip/hip_runtime.h>

// ---------------------------------------------------------------------------
// out = v + SpectralConv1d_time(v)   with T=16, C=8, M=3 modes.
// The rfft->truncate->mix->irfft chain is linear and time-circulant:
//   A[(t',c'),(t,c)] = (1/16) * ( wr[c,c',0]
//        + sum_{m=1,2} 2*( wr[c,c',m]*cos(2*pi*m*D/16) - wi[c,c',m]*sin(2*pi*m*D/16) ) )
//   with D = (t'-t) mod 16   (irfft drops Im of mode 0, matching pocketfft).
// => out = v + A*v : GEMM, M=128, K=128, N = BN*D = 393216 columns.
// Memory-bound: 402 MB traffic -> 17.3 us floor @ 23.3 TB/s.  f16 WMMA with f32
// accumulation; residual added in exact f32.  Data movement uses the CDNA5
// async global<->LDS path (ASYNCcnt) so staging never round-trips VGPRs.
// ---------------------------------------------------------------------------

typedef __attribute__((ext_vector_type(16))) _Float16 v16h;
typedef __attribute__((ext_vector_type(8)))  _Float16 v8h;
typedef __attribute__((ext_vector_type(8)))  float    v8f;
typedef __attribute__((ext_vector_type(4)))  float    v4f;

#define TT      16
#define CC      8
#define KD      128                 // (t,c) flattened
#define BNV     131072
#define DD      3
#define NBATCH  (BNV * DD)          // 393216 columns
#define TILE_B  128                 // batch columns per block
#define SK      136                 // padded LDS row stride (halves) for A
#define SF      136                 // padded LDS row stride (floats) for X

// ---- CDNA5 async global<->LDS movers (ASYNCcnt-tracked, bypass VGPRs) ----
__device__ __forceinline__ void async_ld16(unsigned lds_off, const void* g) {
    asm volatile("global_load_async_to_lds_b128 %0, %1, off"
                 :: "v"(lds_off), "v"((unsigned long long)g) : "memory");
}
__device__ __forceinline__ void async_st16(unsigned lds_off, void* g) {
    asm volatile("global_store_async_from_lds_b128 %1, %0, off"
                 :: "v"(lds_off), "v"((unsigned long long)g) : "memory");
}
__device__ __forceinline__ void wait_async0() {
    asm volatile("s_wait_asynccnt 0x0" ::: "memory");
}
__device__ __forceinline__ void wait_ds0() {
    asm volatile("s_wait_dscnt 0x0" ::: "memory");
}
__device__ __forceinline__ unsigned lds_off(const void* p) {
    // generic pointer to __shared__ has the LDS byte offset in its low 32 bits
    return (unsigned)(unsigned long long)p;
}

// ---- setup: build the 128x128 circulant operator in f16 ----
__global__ void build_A_kernel(const float* __restrict__ wr,
                               const float* __restrict__ wi,
                               _Float16* __restrict__ Aout) {
    const int r  = threadIdx.x;      // output row = t'*8 + c'
    const int tp = r >> 3;
    const int cp = r & 7;
    for (int col = 0; col < KD; ++col) {
        const int t  = col >> 3;
        const int ci = col & 7;
        const int dl = (tp - t) & 15;
        const float* wrp = wr + (ci * CC + cp) * 3;   // w[c_in][c_out][m]
        const float* wip = wi + (ci * CC + cp) * 3;
        float acc = wrp[0];                           // mode 0 (Im dropped by irfft)
        #pragma unroll
        for (int m = 1; m <= 2; ++m) {
            float ang = 6.28318530717958647692f * (float)(m * dl) * (1.0f / 16.0f);
            acc += 2.0f * (wrp[m] * cosf(ang) - wip[m] * sinf(ang));
        }
        Aout[r * KD + col] = (_Float16)(acc * (1.0f / 16.0f));
    }
}

// ---- main: streaming GEMM  out = v + A*v ----
__global__ __launch_bounds__(256) void timeconv_kernel(const float* __restrict__ v,
                                                       const _Float16* __restrict__ Aw,
                                                       float* __restrict__ out) {
    __shared__ _Float16 Ash[KD * SK];        // operator tile (f16), block-shared: 34 KB
    __shared__ float    Xf [TILE_B * SF];    // f32 tile [batch][k]; in-place residual
                                             // and store-transpose buffer: 68 KB

    const int tid  = threadIdx.x;
    const int wave = tid >> 5;
    const int lane = tid & 31;
    const long B0  = (long)blockIdx.x * TILE_B;

    // ---- async stage A (32 KB) cooperatively: 8 x b128 per thread ----
    for (int i = tid; i < (KD * KD) / 8; i += 256) {
        const int row = i >> 4;
        const int ch  = i & 15;
        async_ld16(lds_off(&Ash[row * SK + ch * 8]), Aw + row * KD + ch * 8);
    }

    // ---- async stage X: wave w stages its own batches [16w,16w+16).
    //      Per t the block touches one contiguous 4 KB span (perfect coalescing).
    const int bl  = tid >> 1;           // local batch 0..127
    const int cof = (tid & 1) * 4;      // channel offset 0 or 4
    #pragma unroll
    for (int t = 0; t < TT; ++t) {
        const float* src = v + (size_t)t * NBATCH * CC + (size_t)(B0 + bl) * CC + cof;
        async_ld16(lds_off(&Xf[bl * SF + t * CC + cof]), src);
    }
    wait_async0();      // my async writes to LDS are complete
    __syncthreads();    // everyone else's too

    // ---- per-wave GEMM: 16 batch columns x 128 output rows, K = 128 ----
    const int bg   = wave * 16;
    const int n    = lane & 15;
    const int hsel = lane >> 4;          // 0: lanes 0-15, 1: lanes 16-31

    // B fragments (K x N = 32 x 16): lanes 0-15 hold K = kk*32 + 0..15 of column n,
    // lanes 16-31 hold K = kk*32 + 16..31 (mirrors the documented sparse-B pattern).
    // Convert f32 -> f16 in registers (VALU co-executes with WMMA).
    v16h Bf[4];
    #pragma unroll
    for (int kk = 0; kk < 4; ++kk) {
        const float* p = &Xf[(bg + n) * SF + kk * 32 + hsel * 16];
        v4f a0 = *(const v4f*)p;
        v4f a1 = *(const v4f*)(p + 4);
        v4f a2 = *(const v4f*)(p + 8);
        v4f a3 = *(const v4f*)(p + 12);
        v16h f;
        #pragma unroll
        for (int j = 0; j < 4; ++j) {
            f[j]      = (_Float16)a0[j];
            f[4 + j]  = (_Float16)a1[j];
            f[8 + j]  = (_Float16)a2[j];
            f[12 + j] = (_Float16)a3[j];
        }
        Bf[kk] = f;
    }

    #pragma unroll
    for (int rt = 0; rt < 8; ++rt) {
        v8f acc = {};
        const int row = rt * 16 + n;     // A-fragment M index = lane%16
        #pragma unroll
        for (int kk = 0; kk < 4; ++kk) {
            // A layout (contiguous, matching B): lanes 0-15: K = kb..kb+15;
            // lanes 16-31: K = kb+16..kb+31.  kb = kk*32.
            const _Float16* p = &Ash[row * SK + kk * 32 + hsel * 16];
            v8h lo = *(const v8h*)p;
            v8h hi = *(const v8h*)(p + 8);
            v16h a;
            #pragma unroll
            for (int j = 0; j < 8; ++j) { a[j] = lo[j]; a[8 + j] = hi[j]; }
            acc = __builtin_amdgcn_wmma_f32_16x16x32_f16(false, a, false, Bf[kk],
                                                         (short)0, acc, false, false);
        }
        // residual add in place, vectorized: lane owns the 8 contiguous floats
        // Xf[bg+n][rt*16 + hsel*8 .. +8].  Rows are wave-private -> no barrier.
        float* rp = &Xf[(bg + n) * SF + rt * 16 + hsel * 8];
        v4f r0 = *(const v4f*)rp;
        v4f r1 = *(const v4f*)(rp + 4);
        #pragma unroll
        for (int j = 0; j < 4; ++j) { r0[j] += acc[j]; r1[j] += acc[4 + j]; }
        *(v4f*)rp       = r0;
        *(v4f*)(rp + 4) = r1;
    }

    // ---- async store straight from LDS (coalesced 4 KB/block per t).
    //      DS and ASYNC counters are mutually unordered: drain DS writes first.
    wait_ds0();
    const int sb = bg + (lane >> 1);
    const int sc = (lane & 1) * 4;
    #pragma unroll
    for (int t = 0; t < TT; ++t) {
        float* dst = out + (size_t)t * NBATCH * CC + (size_t)(B0 + sb) * CC + sc;
        async_st16(lds_off(&Xf[sb * SF + t * CC + sc]), dst);
    }
    // S_ENDPGM performs an implicit S_WAIT_IDLE, draining ASYNCcnt.
}

extern "C" void kernel_launch(void* const* d_in, const int* in_sizes, int n_in,
                              void* d_out, int out_size, void* d_ws, size_t ws_size,
                              hipStream_t stream) {
    const float* v  = (const float*)d_in[0];
    const float* wr = (const float*)d_in[1];
    const float* wi = (const float*)d_in[2];
    float* outp     = (float*)d_out;
    _Float16* Aw    = (_Float16*)d_ws;   // 128*128*2 = 32 KB of scratch

    build_A_kernel<<<dim3(1), dim3(128), 0, stream>>>(wr, wi, Aw);
    timeconv_kernel<<<dim3(NBATCH / TILE_B), dim3(256), 0, stream>>>(v, Aw, outp);
}
